// MLA_19043884990750
// MI455X (gfx1250) — matrix operand order repaired and checked by reading, not compiled, not run on hardware
//
#include <hip/hip_runtime.h>
#include <hip/hip_bf16.h>

// ---------------------------------------------------------------------------
// MLA forward for MI455X (gfx1250).
// - All matmuls: v_wmma_f32_16x16x32_bf16, f32 accumulate, bf16 operands
//   staged once (cvt fused into epilogues) -> inner loop is loads + WMMA.
// - B tiles staged to LDS by the Tensor Data Mover (tensor_load_to_lds),
//   double-buffered, synced with s_wait_tensorcnt + workgroup barriers, so the
//   4 waves of a block stop issuing 4x redundant B loads and the DMA overlaps
//   the WMMA stream. Falls back to direct global B loads if the TDM builtin
//   is unavailable.
// Shapes fixed: B=2, S=2048, H=2048, NH=16, QK_NOPE=128, QK_ROPE=64, V=128.
// ---------------------------------------------------------------------------

typedef __attribute__((ext_vector_type(16))) __bf16 v16bf;
typedef __attribute__((ext_vector_type(8)))  __bf16 v8bf;
typedef __attribute__((ext_vector_type(8)))  float  v8f;
typedef unsigned int u32x4 __attribute__((ext_vector_type(4)));
typedef int          i32x4 __attribute__((ext_vector_type(4)));
typedef int          i32x8 __attribute__((ext_vector_type(8)));

#if defined(__has_builtin)
#if __has_builtin(__builtin_amdgcn_tensor_load_to_lds)
#define HAVE_TDM 1
#endif
#endif
#ifndef HAVE_TDM
#define HAVE_TDM 0
#endif

#define DEV __device__ __forceinline__

DEV v8f wmma_bf16(v16bf a, v16bf b, v8f c) {
  return __builtin_amdgcn_wmma_f32_16x16x32_bf16(false, a, false, b, (short)0, c,
                                                 false, false);
}

// ---- fragment loads ----
// A 16x32 (MxK), bf16 row-major. lane<16 -> M=lane, K=0..15;
// lane>=16 -> K=16..31. Contiguous -> two 16B loads.
DEV v16bf load_a_bf(const __bf16* __restrict__ A, long lda, int m0, int k0, int lane) {
  const __bf16* row = A + (long)(m0 + (lane & 15)) * lda + k0 + ((lane >> 4) << 4);
  v8bf lo = *(const v8bf*)row;
  v8bf hi = *(const v8bf*)(row + 8);
  v16bf f;
#pragma unroll
  for (int i = 0; i < 8; ++i) { f[i] = lo[i]; f[8 + i] = hi[i]; }
  return f;
}

// A 16x32 from f32 source (attention probs), cvt on load.
DEV v16bf load_a_f32(const float* __restrict__ A, long lda, int m0, int k0, int lane) {
  const float* row = A + (long)(m0 + (lane & 15)) * lda + k0 + ((lane >> 4) << 4);
  v16bf f;
#pragma unroll
  for (int v = 0; v < 8; ++v) {
    int kp = v << 1;
    f[2 * v]     = (__bf16)row[kp];
    f[2 * v + 1] = (__bf16)row[kp + 1];
  }
  return f;
}

// B 32x16 (KxN) fragment from LDS tile staged as [64 rows][32 K] (BT layout).
// lane<16: N=lane, K=0..15; lane>=16: K=16..31.
DEV v16bf load_bt_lds(const __bf16* sB, int n0l, int lane) {
  const __bf16* row = sB + (n0l + (lane & 15)) * 32 + ((lane >> 4) << 4);
  v8bf lo = *(const v8bf*)row;
  v8bf hi = *(const v8bf*)(row + 8);
  v16bf f;
#pragma unroll
  for (int i = 0; i < 8; ++i) { f[i] = lo[i]; f[8 + i] = hi[i]; }
  return f;
}
// B fragment from LDS tile staged as [32 K][64 N] (non-BT layout).
DEV v16bf load_bn_lds(const __bf16* sB, int n0l, int lane) {
  const __bf16* col = sB + (((lane >> 4) << 4)) * 64 + n0l + (lane & 15);
  v16bf f;
#pragma unroll
  for (int v = 0; v < 8; ++v) {
    f[2 * v]     = col[(2 * v) * 64];
    f[2 * v + 1] = col[(2 * v + 1) * 64];
  }
  return f;
}

#if !HAVE_TDM
// Fallback: B fragments straight from global.
DEV v16bf load_bt_bf(const __bf16* __restrict__ B, long ldb, int n0, int k0, int lane) {
  const __bf16* row = B + (long)(n0 + (lane & 15)) * ldb + k0 + ((lane >> 4) << 4);
  v8bf lo = *(const v8bf*)row;
  v8bf hi = *(const v8bf*)(row + 8);
  v16bf f;
#pragma unroll
  for (int i = 0; i < 8; ++i) { f[i] = lo[i]; f[8 + i] = hi[i]; }
  return f;
}
DEV v16bf load_bn_bf(const __bf16* __restrict__ B, long ldb, int n0, int k0, int lane) {
  const __bf16* col = B + (long)(k0 + ((lane >> 4) << 4)) * ldb + n0 + (lane & 15);
  v16bf f;
#pragma unroll
  for (int v = 0; v < 8; ++v) {
    f[2 * v]     = col[(long)(2 * v) * ldb];
    f[2 * v + 1] = col[(long)(2 * v + 1) * ldb];
  }
  return f;
}
#endif

// C/D 16x16 f32: VGPR v, lane<16 -> M=v, N=lane; lane>=16 -> M=v+8.
DEV void store_c_f32(float* __restrict__ C, long ldc, int m0, int n0, int lane, v8f c) {
  int n  = n0 + (lane & 15);
  int mb = m0 + ((lane >> 4) << 3);
#pragma unroll
  for (int v = 0; v < 8; ++v) C[(long)(mb + v) * ldc + n] = c[v];
}
DEV void store_c_bf(__bf16* __restrict__ C, long ldc, int m0, int n0, int lane, v8f c) {
  int n  = n0 + (lane & 15);
  int mb = m0 + ((lane >> 4) << 3);
#pragma unroll
  for (int v = 0; v < 8; ++v) C[(long)(mb + v) * ldc + n] = (__bf16)c[v];
}

#if HAVE_TDM
// ---- Tensor DMA descriptor build (ISA 8.3/8.4), 2D tile of bf16 ----
DEV u32x4 tdm_g0(unsigned lds_byte_addr, const void* gaddr) {
  unsigned long long ga = (unsigned long long)gaddr;
  u32x4 g;
  g[0] = 1u;                                       // count=1, user D#, no gather
  g[1] = lds_byte_addr;                            // LDS dest (bytes)
  g[2] = (unsigned)(ga & 0xFFFFFFFFu);             // global_addr[31:0]
  g[3] = (unsigned)((ga >> 32) & 0x01FFFFFFu) | (2u << 30);  // addr[56:32], type=2
  return g;
}
DEV i32x8 tdm_g1(unsigned tdim0, unsigned tdim1, unsigned tile0, unsigned tile1,
                 unsigned long long stride0) {
  i32x8 g;
  g[0] = (int)(1u << 16);                          // data_size=1 (2 bytes), mask=0
  g[1] = (int)((tdim0 & 0xFFFFu) << 16);           // tensor_dim0[15:0] @ [63:48]
  g[2] = (int)((tdim0 >> 16) | ((tdim1 & 0xFFFFu) << 16));   // [95:64]
  g[3] = (int)((tdim1 >> 16) | (tile0 << 16));     // tensor_dim1 hi | tile_dim0
  g[4] = (int)(tile1 & 0xFFFFu);                   // tile_dim1 | tile_dim2=0
  g[5] = (int)(stride0 & 0xFFFFFFFFu);             // tensor_dim0_stride[31:0]
  g[6] = (int)((stride0 >> 32) & 0xFFFFu);         // stride0[47:32] | dim1_stride=0
  g[7] = 0;
  return g;
}
DEV void tdm_issue_b(const __bf16* gsrc, unsigned lds_byte_addr, unsigned ldb, bool bt) {
  u32x4 g0 = tdm_g0(lds_byte_addr, gsrc);
  // BT: 64 rows x 32 contiguous K; non-BT: 32 rows x 64 contiguous N.
  i32x8 g1 = bt ? tdm_g1(ldb, 1u << 20, 32u, 64u, ldb)
                : tdm_g1(ldb, 1u << 20, 64u, 32u, ldb);
  i32x4 z4 = {0, 0, 0, 0};
#if __clang_major__ >= 23
  i32x8 z8 = {0, 0, 0, 0, 0, 0, 0, 0};
  __builtin_amdgcn_tensor_load_to_lds(g0, g1, z4, z4, z8, 0);
#else
  __builtin_amdgcn_tensor_load_to_lds(g0, g1, z4, z4, 0);
#endif
}
#endif

// ---- batched GEMM ----
// grid.z -> (b,h) with per-z pointer strides. Block = 128 threads = 4 waves
// stacked along M; wave tile 32x64 (8 WMMA / 32-wide k-step). Block tile
// 128(M) x 64(N). B tile (shared by all 4 waves) staged to LDS via TDM,
// double-buffered.
template <bool BT, bool AF32, bool CB16>
__global__ __launch_bounds__(128) void gemm_wmma(
    const void* __restrict__ Av, const void* __restrict__ Bv, void* __restrict__ Cv,
    int K, int lda, int ldb, int ldc,
    long aB, long aH, long bB, long bH, long cB, long cH, int nH) {
  __shared__ __align__(32) __bf16 sB[2][64 * 32];

  int z  = blockIdx.z;
  int zb = z / nH, zh = z - zb * nH;
  long aOff = (long)zb * aB + (long)zh * aH;
  long bOff = (long)zb * bB + (long)zh * bH;
  long cOff = (long)zb * cB + (long)zh * cH;

  const float*  Af = (const float*)Av + aOff;
  const __bf16* Ab = (const __bf16*)Av + aOff;
  const __bf16* Bb = (const __bf16*)Bv + bOff;

  int lane = threadIdx.x & 31;
  int w    = threadIdx.x >> 5;
  int m0   = blockIdx.y * 128 + w * 32;
  int n0   = blockIdx.x * 64;

  v8f acc[2][4] = {};
  int nsteps = K >> 5;

#if HAVE_TDM
  unsigned ldsb[2];
  ldsb[0] = (unsigned)(unsigned long long)&sB[0][0];  // flat addr low 32 = LDS offset
  ldsb[1] = (unsigned)(unsigned long long)&sB[1][0];
  if (threadIdx.x < 32) {
    const __bf16* g0src = BT ? (Bb + (long)n0 * ldb) : (Bb + n0);
    tdm_issue_b(g0src, ldsb[0], (unsigned)ldb, BT);
  }
#endif

  for (int i = 0; i < nsteps; ++i) {
    int k0 = i << 5;
#if HAVE_TDM
    if (threadIdx.x < 32) {
      if (i + 1 < nsteps) {
        int kn = k0 + 32;
        const __bf16* gsrc = BT ? (Bb + (long)n0 * ldb + kn) : (Bb + (long)kn * ldb + n0);
        tdm_issue_b(gsrc, ldsb[(i + 1) & 1], (unsigned)ldb, BT);
        __builtin_amdgcn_s_wait_tensorcnt(1);   // tile i landed (in-order)
      } else {
        __builtin_amdgcn_s_wait_tensorcnt(0);
      }
    }
    __syncthreads();                            // publish tile i to all waves
    const __bf16* bt = &sB[i & 1][0];
#endif
    if (k0 + 32 < K) {
      if (AF32) __builtin_prefetch(Af + (long)(m0 + (lane & 15)) * lda + k0 + 32, 0, 0);
      else      __builtin_prefetch(Ab + (long)(m0 + (lane & 15)) * lda + k0 + 32, 0, 0);
    }
    v16bf a0, a1;
    if (AF32) { a0 = load_a_f32(Af, lda, m0, k0, lane); a1 = load_a_f32(Af, lda, m0 + 16, k0, lane); }
    else      { a0 = load_a_bf (Ab, lda, m0, k0, lane); a1 = load_a_bf (Ab, lda, m0 + 16, k0, lane); }
    v16bf b[4];
#pragma unroll
    for (int j = 0; j < 4; ++j) {
#if HAVE_TDM
      b[j] = BT ? load_bt_lds(bt, 16 * j, lane) : load_bn_lds(bt, 16 * j, lane);
#else
      b[j] = BT ? load_bt_bf(Bb, ldb, n0 + 16 * j, k0, lane)
                : load_bn_bf(Bb, ldb, n0 + 16 * j, k0, lane);
#endif
    }
#pragma unroll
    for (int j = 0; j < 4; ++j) {
      acc[0][j] = wmma_bf16(a0, b[j], acc[0][j]);
      acc[1][j] = wmma_bf16(a1, b[j], acc[1][j]);
    }
#if HAVE_TDM
    __syncthreads();                            // tile (i) readers done before overwrite
#endif
  }
#pragma unroll
  for (int j = 0; j < 4; ++j) {
    if (CB16) {
      store_c_bf((__bf16*)Cv + cOff, ldc, m0,      n0 + 16 * j, lane, acc[0][j]);
      store_c_bf((__bf16*)Cv + cOff, ldc, m0 + 16, n0 + 16 * j, lane, acc[1][j]);
    } else {
      store_c_f32((float*)Cv + cOff, ldc, m0,      n0 + 16 * j, lane, acc[0][j]);
      store_c_f32((float*)Cv + cOff, ldc, m0 + 16, n0 + 16 * j, lane, acc[1][j]);
    }
  }
}

// ---- elementwise f32 -> bf16 ----
__global__ __launch_bounds__(256) void cvt_kernel(const float* __restrict__ in,
                                                  __bf16* __restrict__ out, long n) {
  long i = ((long)blockIdx.x * 256 + threadIdx.x) * 2;
  if (i < n) { out[i] = (__bf16)in[i]; out[i + 1] = (__bf16)in[i + 1]; }
}

// ---- RMSNorm (f32 in, bf16 out) ----
__global__ __launch_bounds__(256) void rmsnorm_cvt(const float* __restrict__ X,
                                                   const float* __restrict__ w,
                                                   __bf16* __restrict__ Y,
                                                   int L, int strideIn, int strideOut) {
  __shared__ float sred[8];
  const float* x = X + (long)blockIdx.x * strideIn;
  __bf16*      y = Y + (long)blockIdx.x * strideOut;
  float ss = 0.f;
  for (int i = threadIdx.x; i < L; i += 256) { float v = x[i]; ss += v * v; }
  for (int o = 16; o; o >>= 1) ss += __shfl_xor(ss, o, 32);
  if ((threadIdx.x & 31) == 0) sred[threadIdx.x >> 5] = ss;
  __syncthreads();
  if (threadIdx.x < 8) {
    float v = sred[threadIdx.x];
    for (int o = 4; o; o >>= 1) v += __shfl_xor(v, o, 8);
    if (threadIdx.x == 0) sred[0] = v;
  }
  __syncthreads();
  float r = rsqrtf(sred[0] / (float)L + 1e-6f);
  for (int i = threadIdx.x; i < L; i += 256) y[i] = (__bf16)(x[i] * r * w[i]);
}

// ---- RoPE for q: f32 (T,16,192) -> rotated bf16 copy (deinterleave folded) ----
__global__ __launch_bounds__(256) void rope_cvt_q(const float* __restrict__ Q,
                                                  __bf16* __restrict__ Qb) {
  long vec  = (long)blockIdx.x * 8 + (threadIdx.x >> 5);   // (token,head)
  int  lane = threadIdx.x & 31;
  long token = vec >> 4;
  int  h     = (int)(vec & 15);
  const float* in  = Q  + token * 3072 + h * 192;
  __bf16*      out = Qb + token * 3072 + h * 192;
#pragma unroll
  for (int j = 0; j < 4; ++j) out[lane * 4 + j] = (__bf16)in[lane * 4 + j];
  int   pos  = (int)(token & 2047);
  float freq = __powf(10000.0f, -(float)lane / 32.0f);
  float ang  = (float)pos * freq;
  float c = __cosf(ang), s = __sinf(ang);
  float x0 = in[128 + 2 * lane], x1 = in[128 + 2 * lane + 1];
  out[128 + lane] = (__bf16)(x0 * c - x1 * s);
  out[160 + lane] = (__bf16)(x1 * c + x0 * s);
}

// ---- RoPE for k_rope: ckv_f32 cols 512..575 -> bf16 (T,64) ----
__global__ __launch_bounds__(256) void rope_cvt_k(const float* __restrict__ ckv,
                                                  __bf16* __restrict__ kr) {
  long tok  = (long)blockIdx.x * 8 + (threadIdx.x >> 5);
  int  lane = threadIdx.x & 31;
  const float* in  = ckv + tok * 576 + 512;
  __bf16*      out = kr  + tok * 64;
  int   pos  = (int)(tok & 2047);
  float freq = __powf(10000.0f, -(float)lane / 32.0f);
  float ang  = (float)pos * freq;
  float c = __cosf(ang), s = __sinf(ang);
  float x0 = in[2 * lane], x1 = in[2 * lane + 1];
  out[lane]      = (__bf16)(x0 * c - x1 * s);
  out[32 + lane] = (__bf16)(x1 * c + x0 * s);
}

// ---- Kc[b,h,s,0:192] = concat(k_nope[b,s,h,:], k_rope[b,s,:]) (bf16) ----
__global__ __launch_bounds__(256) void build_k_kernel(const __bf16* __restrict__ kvup,
                                                      const __bf16* __restrict__ kr,
                                                      __bf16* __restrict__ Kc) {
  long i = (long)blockIdx.x * 256 + threadIdx.x;   // total = 2*16*2048*192
  int  d = (int)(i % 192);
  long t = i / 192;
  int  s = (int)(t & 2047);
  long bh = t >> 11;
  int  h = (int)(bh & 15);
  int  b = (int)(bh >> 4);
  long tok = (long)b * 2048 + s;
  Kc[i] = (d < 128) ? kvup[tok * 4096 + h * 256 + d] : kr[tok * 64 + (d - 128)];
}

// ---- causal softmax in place over rows of 2048 (f32, in d_out) ----
__global__ __launch_bounds__(256) void softmax_causal(float* __restrict__ attn,
                                                      float scale) {
  __shared__ float sred[8];
  long row = blockIdx.x;
  int  q   = (int)(row & 2047);
  float* x = attn + row * 2048;
  int n = q + 1;

  float m = -3.4e38f;
  for (int i = threadIdx.x; i < n; i += 256) m = fmaxf(m, x[i]);
  for (int o = 16; o; o >>= 1) m = fmaxf(m, __shfl_xor(m, o, 32));
  if ((threadIdx.x & 31) == 0) sred[threadIdx.x >> 5] = m;
  __syncthreads();
  if (threadIdx.x < 8) {
    float v = sred[threadIdx.x];
    for (int o = 4; o; o >>= 1) v = fmaxf(v, __shfl_xor(v, o, 8));
    if (threadIdx.x == 0) sred[0] = v;
  }
  __syncthreads();
  float bm = sred[0] * scale;
  __syncthreads();

  float sum = 0.f;
  for (int i = threadIdx.x; i < n; i += 256) {
    float e = __expf(x[i] * scale - bm);
    x[i] = e;
    sum += e;
  }
  for (int o = 16; o; o >>= 1) sum += __shfl_xor(sum, o, 32);
  if ((threadIdx.x & 31) == 0) sred[threadIdx.x >> 5] = sum;
  __syncthreads();
  if (threadIdx.x < 8) {
    float v = sred[threadIdx.x];
    for (int o = 4; o; o >>= 1) v += __shfl_xor(v, o, 8);
    if (threadIdx.x == 0) sred[0] = v;
  }
  __syncthreads();
  float rs = 1.0f / sred[0];
  for (int i = threadIdx.x; i < n; i += 256) x[i] *= rs;
  for (int i = n + threadIdx.x; i < 2048; i += 256) x[i] = 0.f;
}

extern "C" void kernel_launch(void* const* d_in, const int* in_sizes, int n_in,
                              void* d_out, int out_size, void* d_ws, size_t ws_size,
                              hipStream_t stream) {
  (void)in_sizes; (void)n_in; (void)out_size; (void)ws_size;
  const float* hidden    = (const float*)d_in[0];
  const float* w_q_down  = (const float*)d_in[3];
  const float* q_norm_w  = (const float*)d_in[4];
  const float* w_q_up    = (const float*)d_in[5];
  const float* w_kv_down = (const float*)d_in[6];
  const float* kv_norm_w = (const float*)d_in[7];
  const float* w_kv_up   = (const float*)d_in[8];
  const float* w_out     = (const float*)d_in[9];

  constexpr long Bb = 2, S = 2048, Hm = 2048, NH = 16, T = Bb * S;

  char* p = (char*)d_ws;
  float*  q_lat_f = (float*)p;  p += T * 1536 * 4;
  float*  q_f     = (float*)p;  p += T * 3072 * 4;
  float*  ckv_f   = (float*)p;  p += T * 576 * 4;
  __bf16* hid_b   = (__bf16*)p; p += T * 2048 * 2;
  __bf16* wqd_b   = (__bf16*)p; p += 1536L * 2048 * 2;
  __bf16* wqu_b   = (__bf16*)p; p += 3072L * 1536 * 2;
  __bf16* wkvd_b  = (__bf16*)p; p += 576L * 2048 * 2;
  __bf16* wkvu_b  = (__bf16*)p; p += 4096L * 512 * 2;
  __bf16* wout_b  = (__bf16*)p; p += 2048L * 2048 * 2;
  __bf16* qlat_b  = (__bf16*)p; p += T * 1536 * 2;
  __bf16* q_b     = (__bf16*)p; p += T * 3072 * 2;
  __bf16* ckv_b   = (__bf16*)p; p += T * 512 * 2;
  __bf16* kr_b    = (__bf16*)p; p += T * 64 * 2;
  __bf16* kvup_b  = (__bf16*)p; p += T * 4096 * 2;
  __bf16* Kc_b    = (__bf16*)p; p += Bb * NH * S * 192 * 2;
  __bf16* av_b    = (__bf16*)p; p += T * 2048 * 2;

  float* outp = (float*)d_out;
  float* attn = outp + T * Hm;

  dim3 blk(128);
  const float scale = 0.072168783648703220563f;   // 1/sqrt(192)

  cvt_kernel<<<(int)(T * 2048 / 512), 256, 0, stream>>>(hidden, hid_b, T * 2048);
  cvt_kernel<<<(int)(1536L * 2048 / 512), 256, 0, stream>>>(w_q_down, wqd_b, 1536L * 2048);
  cvt_kernel<<<(int)(3072L * 1536 / 512), 256, 0, stream>>>(w_q_up, wqu_b, 3072L * 1536);
  cvt_kernel<<<(int)(576L * 2048 / 512), 256, 0, stream>>>(w_kv_down, wkvd_b, 576L * 2048);
  cvt_kernel<<<(int)(4096L * 512 / 512), 256, 0, stream>>>(w_kv_up, wkvu_b, 4096L * 512);
  cvt_kernel<<<(int)(2048L * 2048 / 512), 256, 0, stream>>>(w_out, wout_b, 2048L * 2048);

  gemm_wmma<true, false, false><<<dim3(1536 / 64, T / 128, 1), blk, 0, stream>>>(
      hid_b, wqd_b, q_lat_f, 2048, 2048, 2048, 1536, 0, 0, 0, 0, 0, 0, 1);
  rmsnorm_cvt<<<(int)T, 256, 0, stream>>>(q_lat_f, q_norm_w, qlat_b, 1536, 1536, 1536);
  gemm_wmma<true, false, false><<<dim3(3072 / 64, T / 128, 1), blk, 0, stream>>>(
      qlat_b, wqu_b, q_f, 1536, 1536, 1536, 3072, 0, 0, 0, 0, 0, 0, 1);
  rope_cvt_q<<<(int)(T * NH / 8), 256, 0, stream>>>(q_f, q_b);
  gemm_wmma<true, false, false><<<dim3(576 / 64, T / 128, 1), blk, 0, stream>>>(
      hid_b, wkvd_b, ckv_f, 2048, 2048, 2048, 576, 0, 0, 0, 0, 0, 0, 1);
  rope_cvt_k<<<(int)(T / 8), 256, 0, stream>>>(ckv_f, kr_b);
  rmsnorm_cvt<<<(int)T, 256, 0, stream>>>(ckv_f, kv_norm_w, ckv_b, 512, 576, 512);
  gemm_wmma<true, false, true><<<dim3(4096 / 64, T / 128, 1), blk, 0, stream>>>(
      ckv_b, wkvu_b, kvup_b, 512, 512, 512, 4096, 0, 0, 0, 0, 0, 0, 1);
  build_k_kernel<<<(int)(Bb * NH * S * 192 / 256), 256, 0, stream>>>(kvup_b, kr_b, Kc_b);
  gemm_wmma<true, false, false><<<dim3(2048 / 64, 2048 / 128, Bb * NH), blk, 0, stream>>>(
      q_b, Kc_b, attn, 192, 3072, 192, 2048,
      S * 3072, 192, NH * S * 192, S * 192, NH * S * S, S * S, (int)NH);
  softmax_causal<<<(int)(Bb * NH * S), 256, 0, stream>>>(attn, scale);
  gemm_wmma<false, true, true><<<dim3(128 / 64, 2048 / 128, Bb * NH), blk, 0, stream>>>(
      attn, kvup_b + 128, av_b, 2048, 2048, 4096, 2048,
      NH * S * S, S * S, S * 4096, 256, S * 2048, 128, (int)NH);
  gemm_wmma<true, false, false><<<dim3(2048 / 64, T / 128, 1), blk, 0, stream>>>(
      av_b, wout_b, outp, 2048, 2048, 2048, 2048, 0, 0, 0, 0, 0, 0, 1);
}
